// BipartiteGraph1d_86346022518991
// MI455X (gfx1250) — compile-verified, matrix operand order checked by reading, not run on hardware
//
#include <hip/hip_runtime.h>

typedef __attribute__((ext_vector_type(8)))  float  v8f;
typedef __attribute__((ext_vector_type(8)))  __bf16 v8bf;
typedef __attribute__((ext_vector_type(16))) __bf16 v16bf;
typedef __attribute__((ext_vector_type(4)))  int    v4i;
typedef __attribute__((address_space(1))) v4i* gas_v4i;   // global-AS int4*
typedef __attribute__((address_space(3))) v4i* las_v4i;   // LDS-AS int4*

#define NROWS 16384
#define DDIM  1024
#define BM 128
#define BN 128
#define BK 32
#define LDK 40   // bf16 elements per LDS row (32 + 8 pad): keeps 16B alignment, banks spread

#if __has_builtin(__builtin_amdgcn_global_load_async_to_lds_b128)
#define ASYNC_G2L 1
#else
#define ASYNC_G2L 0
#endif

struct LayerArgs {
    const float* A[7];            // history[j]
    const unsigned short* BH;     // pre-split K-major bf16 hi panels, j-th at +j*D*D
    const unsigned short* BL;     // pre-split K-major bf16 lo panels
    int   nj;
    float inv_nj;
};

struct PrepArgs {
    const float* Batt[7];
    const float* Bmul[7];
};

// Split two fp32 values (a,b) into packed bf16 hi parts and bf16 lo parts.
// hi = truncate-to-bf16 (exact, so a - hi is exactly representable);
// lo = round-to-bf16 of the remainder. Packed as (a in [15:0], b in [31:16]).
__device__ __forceinline__ void split2(float a, float b, unsigned& hi, unsigned& lo)
{
    const unsigned ua = __float_as_uint(a);
    const unsigned ub = __float_as_uint(b);
    const unsigned ha = ua & 0xFFFF0000u;
    const unsigned hb = ub & 0xFFFF0000u;
    const float    ra = a - __uint_as_float(ha);
    const float    rb = b - __uint_as_float(hb);
    hi = (ha >> 16) | hb;
    const unsigned la = (__float_as_uint(ra) + 0x8000u) >> 16;
    const unsigned lb = (__float_as_uint(rb) + 0x8000u) >> 16;
    lo = la | (lb << 16);
}

// 16-byte global->LDS copy: async-to-LDS engine if available, else through VGPRs.
__device__ __forceinline__ void copy16_g2l(const unsigned short* g, unsigned short* l)
{
#if ASYNC_G2L
    __builtin_amdgcn_global_load_async_to_lds_b128((gas_v4i)g, (las_v4i)l, 0, 0);
#else
    *reinterpret_cast<uint4*>(l) = *reinterpret_cast<const uint4*>(g);
#endif
}

__device__ __forceinline__ void wait_async_copies()
{
#if ASYNC_G2L
#if __has_builtin(__builtin_amdgcn_s_wait_asynccnt)
    __builtin_amdgcn_s_wait_asynccnt(0);
#else
    asm volatile("s_wait_asynccnt 0" ::: "memory");
#endif
#endif
}

// Load a 16x32 bf16 WMMA operand fragment from a K-major LDS row.
// Lane layout (ISA 7.12.2, 16-bit A 16x32): row = lane%16, half = lane/16,
// K = {8*half .. 8*half+7} and {16+8*half .. 16+8*half+7}  -> two b128 loads.
__device__ __forceinline__ v16bf load_frag(const unsigned short* rowbase, int half)
{
    v16bf r;
    ((v8bf*)&r)[0] = *(const v8bf*)(rowbase + 8 * half);
    ((v8bf*)&r)[1] = *(const v8bf*)(rowbase + 16 + 8 * half);
    return r;
}

// Per-layer prep: effective B_j = Batt_j (* Bmul_j), transposed to K-major and
// split to bf16 hi/lo.  outH/outL[j][n*D + k] = split(B_j[k][n]).
__global__ __launch_bounds__(256) void prep_b(PrepArgs pa,
                                              unsigned short* __restrict__ outH,
                                              unsigned short* __restrict__ outL)
{
    __shared__ float tile[64][65];           // [k_local][n_local], +1 pad
    const int j = blockIdx.z;
    const float* __restrict__ Batt = pa.Batt[j];
    const float* __restrict__ Bmul = pa.Bmul[j];
    unsigned short* __restrict__ oh = outH + (size_t)j * DDIM * DDIM;
    unsigned short* __restrict__ ol = outL + (size_t)j * DDIM * DDIM;

    const int kb = blockIdx.x * 64;
    const int nb = blockIdx.y * 64;
    const int t  = threadIdx.x;
    const int r  = t >> 4;                   // 0..15
    const int c4 = (t & 15) * 4;             // 0..60

    #pragma unroll
    for (int it = 0; it < 4; ++it) {
        const int kl = r + 16 * it;
        const size_t off = (size_t)(kb + kl) * DDIM + nb + c4;
        float4 v = *reinterpret_cast<const float4*>(Batt + off);
        if (Bmul) {
            const float4 m = *reinterpret_cast<const float4*>(Bmul + off);
            v.x *= m.x; v.y *= m.y; v.z *= m.z; v.w *= m.w;
        }
        tile[kl][c4 + 0] = v.x;
        tile[kl][c4 + 1] = v.y;
        tile[kl][c4 + 2] = v.z;
        tile[kl][c4 + 3] = v.w;
    }
    __syncthreads();
    #pragma unroll
    for (int it = 0; it < 4; ++it) {
        const int nl = r + 16 * it;
        unsigned h0, l0, h1, l1;
        split2(tile[c4 + 0][nl], tile[c4 + 1][nl], h0, l0);
        split2(tile[c4 + 2][nl], tile[c4 + 3][nl], h1, l1);
        const size_t off = (size_t)(nb + nl) * DDIM + kb + c4;
        *reinterpret_cast<unsigned*>(oh + off)     = h0;
        *reinterpret_cast<unsigned*>(oh + off + 2) = h1;
        *reinterpret_cast<unsigned*>(ol + off)     = l0;
        *reinterpret_cast<unsigned*>(ol + off + 2) = l1;
    }
}

// C_tile = (1/nj) * sum_j A_j @ B_j for a 128x128 output tile.
// fp32 emulated as bf16x3: Ah*Bh + Ah*Bl + Al*Bh, fp32 accumulation.
__global__ __launch_bounds__(256) void gemm_layer(LayerArgs args, float* __restrict__ out)
{
    __shared__ unsigned short Ah[BM * LDK];
    __shared__ unsigned short Al[BM * LDK];
    __shared__ unsigned short Bh[BN * LDK];   // K-major: Bh[n*LDK + k]
    __shared__ unsigned short Bl[BN * LDK];

    const int t    = threadIdx.x;
    const int lane = t & 31;
    const int wave = t >> 5;          // 0..7
    const int wm   = wave & 1;        // M-split: 2 waves
    const int wn   = wave >> 1;       // N-split: 4 waves
    const int half = lane >> 4;       // 0 or 1
    const int lrow = lane & 15;

    const int rowBase = blockIdx.x * BM;
    const int colBase = blockIdx.y * BN;

    v8f acc[4][2];
    #pragma unroll
    for (int mt = 0; mt < 4; ++mt)
        #pragma unroll
        for (int nt = 0; nt < 2; ++nt)
            acc[mt][nt] = (v8f){0.f, 0.f, 0.f, 0.f, 0.f, 0.f, 0.f, 0.f};

    for (int j = 0; j < args.nj; ++j) {
        const float* __restrict__ A = args.A[j];
        const unsigned short* __restrict__ bh = args.BH + (size_t)j * DDIM * DDIM;
        const unsigned short* __restrict__ bl = args.BL + (size_t)j * DDIM * DDIM;

        for (int k0 = 0; k0 < DDIM; k0 += BK) {
            __syncthreads();

            // ---- stage B tile: direct 16B copies of pre-split K-major bf16 ----
            {
                #pragma unroll
                for (int it = 0; it < 2; ++it) {
                    const int task = t + 256 * it;       // 0..511
                    const int n = task >> 2;             // 0..127
                    const int c = (task & 3) * 8;        // 0,8,16,24 (bf16 elems)
                    const size_t goff = (size_t)(colBase + n) * DDIM + k0 + c;
                    const int loff = n * LDK + c;
                    copy16_g2l(bh + goff, &Bh[loff]);
                    copy16_g2l(bl + goff, &Bl[loff]);
                }
            }
            // ---- stage A tile: 128 x 32 fp32 -> hi/lo bf16 (K-major rows) ----
            {
                const int r  = t >> 3;           // 0..31
                const int kc = (t & 7) * 4;      // 0..28
                #pragma unroll
                for (int it = 0; it < 4; ++it) {
                    const int rr = r + 32 * it;
                    const float4 v = *reinterpret_cast<const float4*>(
                        A + (size_t)(rowBase + rr) * DDIM + k0 + kc);
                    unsigned h0, l0, h1, l1;
                    split2(v.x, v.y, h0, l0);
                    split2(v.z, v.w, h1, l1);
                    *reinterpret_cast<unsigned*>(&Ah[rr * LDK + kc])     = h0;
                    *reinterpret_cast<unsigned*>(&Ah[rr * LDK + kc + 2]) = h1;
                    *reinterpret_cast<unsigned*>(&Al[rr * LDK + kc])     = l0;
                    *reinterpret_cast<unsigned*>(&Al[rr * LDK + kc + 2]) = l1;
                }
            }
            wait_async_copies();
            __syncthreads();

            // ---- compute: 3 bf16 WMMAs per 16x16 tile cover fp32 K=32 ----
            v16bf afh[4], bfh[2];
            #pragma unroll
            for (int mt = 0; mt < 4; ++mt)
                afh[mt] = load_frag(&Ah[(64 * wm + 16 * mt + lrow) * LDK], half);
            #pragma unroll
            for (int nt = 0; nt < 2; ++nt)
                bfh[nt] = load_frag(&Bh[(32 * wn + 16 * nt + lrow) * LDK], half);

            #pragma unroll
            for (int mt = 0; mt < 4; ++mt)
                #pragma unroll
                for (int nt = 0; nt < 2; ++nt)
                    acc[mt][nt] = __builtin_amdgcn_wmma_f32_16x16x32_bf16(
                        false, afh[mt], false, bfh[nt],
                        (short)0, acc[mt][nt], false, false);

            {   // Ah x Bl
                v16bf bfl[2];
                #pragma unroll
                for (int nt = 0; nt < 2; ++nt)
                    bfl[nt] = load_frag(&Bl[(32 * wn + 16 * nt + lrow) * LDK], half);
                #pragma unroll
                for (int mt = 0; mt < 4; ++mt)
                    #pragma unroll
                    for (int nt = 0; nt < 2; ++nt)
                        acc[mt][nt] = __builtin_amdgcn_wmma_f32_16x16x32_bf16(
                            false, afh[mt], false, bfl[nt],
                            (short)0, acc[mt][nt], false, false);
            }
            {   // Al x Bh
                v16bf afl[4];
                #pragma unroll
                for (int mt = 0; mt < 4; ++mt)
                    afl[mt] = load_frag(&Al[(64 * wm + 16 * mt + lrow) * LDK], half);
                #pragma unroll
                for (int mt = 0; mt < 4; ++mt)
                    #pragma unroll
                    for (int nt = 0; nt < 2; ++nt)
                        acc[mt][nt] = __builtin_amdgcn_wmma_f32_16x16x32_bf16(
                            false, afl[mt], false, bfh[nt],
                            (short)0, acc[mt][nt], false, false);
            }
        }
    }

    // ---- store (scaled by 1/nj); C/D layout: VGPR r -> M = r + 8*half, N = lrow ----
    #pragma unroll
    for (int mt = 0; mt < 4; ++mt)
        #pragma unroll
        for (int nt = 0; nt < 2; ++nt)
            #pragma unroll
            for (int r = 0; r < 8; ++r) {
                const int row = rowBase + 64 * wm + 16 * mt + r + 8 * half;
                const int col = colBase + 32 * wn + 16 * nt + lrow;
                out[(size_t)row * DDIM + col] = acc[mt][nt][r] * args.inv_nj;
            }
}

// In-place ReLU + LayerNorm (biased variance, eps=1e-5) over a 1024-wide row.
__global__ __launch_bounds__(256) void relu_ln(float* __restrict__ buf)
{
    const int row = blockIdx.x;
    float* p = buf + (size_t)row * DDIM;
    const int t = threadIdx.x;

    float v[4];
    float s = 0.f, s2 = 0.f;
    #pragma unroll
    for (int i = 0; i < 4; ++i) {
        float x = p[t + 256 * i];
        x = fmaxf(x, 0.f);
        v[i] = x;
        s += x;
        s2 += x * x;
    }
    #pragma unroll
    for (int off = 16; off > 0; off >>= 1) {
        s  += __shfl_xor(s,  off, 32);
        s2 += __shfl_xor(s2, off, 32);
    }
    __shared__ float sh_s[8], sh_s2[8];
    const int lane = t & 31, wave = t >> 5;
    if (lane == 0) { sh_s[wave] = s; sh_s2[wave] = s2; }
    __syncthreads();
    if (wave == 0) {
        float a = (lane < 8) ? sh_s[lane]  : 0.f;
        float b = (lane < 8) ? sh_s2[lane] : 0.f;
        #pragma unroll
        for (int off = 4; off > 0; off >>= 1) {
            a += __shfl_xor(a, off, 32);
            b += __shfl_xor(b, off, 32);
        }
        if (lane == 0) { sh_s[0] = a; sh_s2[0] = b; }
    }
    __syncthreads();
    const float mean = sh_s[0] * (1.f / (float)DDIM);
    const float var  = sh_s2[0] * (1.f / (float)DDIM) - mean * mean;
    const float rstd = rsqrtf(var + 1e-5f);
    #pragma unroll
    for (int i = 0; i < 4; ++i)
        p[t + 256 * i] = (v[i] - mean) * rstd;
}

extern "C" void kernel_launch(void* const* d_in, const int* in_sizes, int n_in,
                              void* d_out, int out_size, void* d_ws, size_t ws_size,
                              hipStream_t stream)
{
    const float* x     = (const float*)d_in[0];
    const float* stt   = (const float*)d_in[1];  // source_to_target (D,D)
    const float* tts   = (const float*)d_in[2];  // target_to_source (D,D)
    const float* att_s = (const float*)d_in[3];  // (8,D,D)
    const float* att_t = (const float*)d_in[4];  // (8,D,D)
    const int num_layers = 8;                    // fixed by setup_inputs

    float* out = (float*)d_out;
    float* ws  = (float*)d_ws;

    float* hist[8];
    hist[0] = const_cast<float*>(x);
    for (int i = 1; i <= 5; ++i)
        hist[i] = ws + (size_t)(i - 1) * NROWS * DDIM;
    hist[6] = out;                               // history[-2]
    hist[7] = out + (size_t)NROWS * DDIM;        // history[-1]

    // B-panel pool right after the history buffers: 7 hi + 7 lo panels of D*D bf16
    unsigned short* bpoolH = (unsigned short*)(ws + (size_t)5 * NROWS * DDIM);
    unsigned short* bpoolL = bpoolH + (size_t)7 * DDIM * DDIM;

    const dim3 grid(NROWS / BM, DDIM / BN);
    const dim3 block(256);

    for (int i = 1; i < num_layers; ++i) {
        PrepArgs pa{};
        LayerArgs la{};
        la.nj = i;
        la.inv_nj = 1.0f / (float)i;
        la.BH = bpoolH;
        la.BL = bpoolL;
        for (int j = 0; j < i; ++j) {
            const int k = i - j - 1;             // k < 8, so k % 8 == k
            la.A[j] = hist[j];
            if ((i & 1) == 0) {
                pa.Batt[j] = att_s + (size_t)k * DDIM * DDIM;
                pa.Bmul[j] = ((j & 1) == 0) ? nullptr : tts;
            } else {
                pa.Batt[j] = att_t + (size_t)k * DDIM * DDIM;
                pa.Bmul[j] = ((j & 1) == 0) ? stt : nullptr;
            }
        }
        prep_b<<<dim3(DDIM / 64, DDIM / 64, i), block, 0, stream>>>(pa, bpoolH, bpoolL);
        gemm_layer<<<grid, block, 0, stream>>>(la, hist[i]);
        relu_ln<<<NROWS, block, 0, stream>>>(hist[i]);
    }
}
